// GlobalAttention_13305808683355
// MI455X (gfx1250) — matrix operand order repaired and checked by reading, not compiled, not run on hardware
//
#include <hip/hip_runtime.h>
#include <hip/hip_bf16.h>

typedef __attribute__((ext_vector_type(16))) _Float16 v16h;
typedef __attribute__((ext_vector_type(8)))  _Float16 v8h;
typedef __attribute__((ext_vector_type(8)))  float    v8f;

#define C_DIM   512
#define N3C     1536
#define HEADS   8
#define HDIM    64
#define SEQ     2048
#define BATCH   4
#define ROWS    8192            // BATCH*SEQ
#define SCALE   0.125f          // 64^-0.5
#define NEG_INF (-1e30f)

__device__ __forceinline__ v8f zero8() {
    v8f z;
#pragma unroll
    for (int i = 0; i < 8; ++i) z[i] = 0.0f;
    return z;
}

__device__ __forceinline__ v16h combine16(v8h lo, v8h hi) {
    v16h r;
#pragma unroll
    for (int i = 0; i < 8; ++i) { r[i] = lo[i]; r[i + 8] = hi[i]; }
    return r;
}

// A-matrix 16x32 f16 fragment from row-major [row][ld] source.
__device__ __forceinline__ v16h load_a_frag(const _Float16* base, int ld, int k0, int lane) {
    const int row = lane & 15, lh = lane >> 4;
    const _Float16* p = base + (size_t)row * ld + k0 + lh * 8;
    return combine16(*(const v8h*)p, *(const v8h*)(p + 16));
}

// B-matrix 32x16 f16 fragment from K-contiguous source wT[col][k].
__device__ __forceinline__ v16h load_b_frag(const _Float16* base, int ld, int k0, int lane) {
    const int col = lane & 15, lh = lane >> 4;
    const _Float16* p = base + (size_t)col * ld + k0 + lh * 8;
    return combine16(*(const v8h*)p, *(const v8h*)(p + 16));
}

__device__ __forceinline__ v8f wmma16(v16h a, v16h b, v8f c) {
    return __builtin_amdgcn_wmma_f32_16x16x32_f16(false, a, false, b, (short)0, c, false, false);
}

// ---------------------------------------------------------------- LayerNorm
__global__ void __launch_bounds__(256) ln_kernel(const float* __restrict__ x,
                                                 const float* __restrict__ gamma,
                                                 const float* __restrict__ beta,
                                                 _Float16* __restrict__ xn) {
    const int row = blockIdx.x, tid = threadIdx.x;
    const float* xr = x + (size_t)row * C_DIM;
    float2 v = ((const float2*)xr)[tid];
    __shared__ float ss[256], sq[256];
    ss[tid] = v.x + v.y;
    sq[tid] = v.x * v.x + v.y * v.y;
    __syncthreads();
#pragma unroll
    for (int off = 128; off > 0; off >>= 1) {
        if (tid < off) { ss[tid] += ss[tid + off]; sq[tid] += sq[tid + off]; }
        __syncthreads();
    }
    const float mean = ss[0] * (1.0f / C_DIM);
    const float var  = sq[0] * (1.0f / C_DIM) - mean * mean;
    const float rstd = rsqrtf(var + 1e-5f);
    const int c = tid * 2;
    _Float16* o = xn + (size_t)row * C_DIM + c;
    o[0] = (_Float16)((v.x - mean) * rstd * gamma[c]     + beta[c]);
    o[1] = (_Float16)((v.y - mean) * rstd * gamma[c + 1] + beta[c + 1]);
}

// ------------------------------------------- f32 [K][N] -> f16 transposed [N][K]
__global__ void __launch_bounds__(256) wconv_kernel(const float* __restrict__ w,
                                                    _Float16* __restrict__ wT,
                                                    int K, int N) {
    const size_t t = (size_t)blockIdx.x * blockDim.x + threadIdx.x;
    if (t >= (size_t)K * N) return;
    const int k = (int)(t % K), n = (int)(t / K);
    wT[t] = (_Float16)w[(size_t)k * N + n];
}

// ---------------------------------------------------------------- GEMM helpers
__device__ __forceinline__ void gemm_load(const _Float16* __restrict__ ar0,
                                          const _Float16* __restrict__ ar1,
                                          const _Float16* __restrict__ wcol,
                                          int k, int lane,
                                          v16h& a0, v16h& a1, v16h (&b)[4]) {
    a0 = load_a_frag(ar0, C_DIM, k, lane);
    a1 = load_a_frag(ar1, C_DIM, k, lane);
#pragma unroll
    for (int t = 0; t < 4; ++t)
        b[t] = load_b_frag(wcol + (size_t)(t * 16) * C_DIM, C_DIM, k, lane);
}

__device__ __forceinline__ void gemm_mma(v8f (&acc)[2][4],
                                         const v16h& a0, const v16h& a1, const v16h (&b)[4]) {
#pragma unroll
    for (int t = 0; t < 4; ++t) {
        acc[0][t] = wmma16(a0, b[t], acc[0][t]);
        acc[1][t] = wmma16(a1, b[t], acc[1][t]);
    }
}

// ---------------------------------------------------------------- QKV GEMM
// Wave tile 32x64; 2-deep software pipeline with register ping-pong (no copies).
__global__ void __launch_bounds__(256) qkv_gemm(const _Float16* __restrict__ xn,
                                                const _Float16* __restrict__ wT,
                                                _Float16* __restrict__ qb,
                                                _Float16* __restrict__ kb,
                                                _Float16* __restrict__ vtb) {
    const int lane = threadIdx.x & 31, wave = threadIdx.x >> 5;
    const int rowBase = blockIdx.y * 256 + wave * 32;
    const int colBase = blockIdx.x * 64;

    v8f acc[2][4];
#pragma unroll
    for (int rb = 0; rb < 2; ++rb)
#pragma unroll
        for (int t = 0; t < 4; ++t) acc[rb][t] = zero8();

    const _Float16* ar0  = xn + (size_t)rowBase * C_DIM;
    const _Float16* ar1  = ar0 + (size_t)16 * C_DIM;
    const _Float16* wcol = wT + (size_t)colBase * C_DIM;

    v16h a0A, a1A, bA[4], a0B, a1B, bB[4];
    gemm_load(ar0, ar1, wcol, 0, lane, a0A, a1A, bA);

    for (int k0 = 0; k0 < C_DIM; k0 += 64) {
        gemm_load(ar0, ar1, wcol, k0 + 32, lane, a0B, a1B, bB);
        gemm_mma(acc, a0A, a1A, bA);
        gemm_load(ar0, ar1, wcol, (k0 + 64) & (C_DIM - 1), lane, a0A, a1A, bA);
        gemm_mma(acc, a0B, a1B, bB);
    }

    const int col = lane & 15, lh = lane >> 4;
#pragma unroll
    for (int rb = 0; rb < 2; ++rb) {
#pragma unroll
        for (int t = 0; t < 4; ++t) {
#pragma unroll
            for (int r = 0; r < 8; ++r) {
                const int gm = rowBase + rb * 16 + r + 8 * lh;
                const int gc = colBase + t * 16 + col;
                const float val = acc[rb][t][r];
                const int bb = gm >> 11, nseq = gm & 2047;
                const int sec = gc >> 9, cc = gc & 511, h = cc >> 6, d = cc & 63;
                const size_t bh = (size_t)(bb * HEADS + h);
                if (sec == 0)      qb[(bh * SEQ + nseq) * HDIM + d] = (_Float16)(val * SCALE);
                else if (sec == 1) kb[(bh * SEQ + nseq) * HDIM + d] = (_Float16)val;
                else               vtb[(bh * HDIM + d) * SEQ + nseq] = (_Float16)val;
            }
        }
    }
}

// ------------------------------------------------------- Flash attention core
// One 32-key block: 4 S-WMMAs on current K frags, prefetch next K frags (written
// directly into the other buffer set), prefetch V frags, online softmax, LDS
// transpose of P, 4 O-WMMAs.
__device__ __forceinline__ void attn_block(const _Float16* __restrict__ kp,
                                           const _Float16* __restrict__ vtp,
                                           _Float16* myP,
                                           int m0, int mn, int lane, int col, int lh,
                                           const v16h& qf0, const v16h& qf1,
                                           const v16h& kf0, const v16h& kf1,
                                           const v16h& kf2, const v16h& kf3,
                                           v16h& nk0, v16h& nk1, v16h& nk2, v16h& nk3,
                                           v8f& o0, v8f& o1, v8f& o2, v8f& o3,
                                           float (&m_r)[8], float (&l_r)[8]) {
    v8f s0 = zero8(), s1 = zero8();
    s0 = wmma16(qf0, kf0, s0);
    s0 = wmma16(qf1, kf1, s0);
    s1 = wmma16(qf0, kf2, s1);
    s1 = wmma16(qf1, kf3, s1);

    // next key block's K fragments -> alternate buffer set (no register copies)
    nk0 = load_b_frag(kp + (size_t)mn * HDIM,        HDIM, 0,  lane);
    nk1 = load_b_frag(kp + (size_t)mn * HDIM,        HDIM, 32, lane);
    nk2 = load_b_frag(kp + (size_t)(mn + 16) * HDIM, HDIM, 0,  lane);
    nk3 = load_b_frag(kp + (size_t)(mn + 16) * HDIM, HDIM, 32, lane);
    // this block's V fragments (consumed after the LDS round-trip)
    const v16h vf0 = load_b_frag(vtp,                    SEQ, m0, lane);
    const v16h vf1 = load_b_frag(vtp + (size_t)16 * SEQ, SEQ, m0, lane);
    const v16h vf2 = load_b_frag(vtp + (size_t)32 * SEQ, SEQ, m0, lane);
    const v16h vf3 = load_b_frag(vtp + (size_t)48 * SEQ, SEQ, m0, lane);

    // online softmax (row stats replicated across each 16-lane half)
    float alpha[8];
#pragma unroll
    for (int r = 0; r < 8; ++r) {
        float mx = fmaxf(s0[r], s1[r]);
#pragma unroll
        for (int msk = 1; msk < 16; msk <<= 1) mx = fmaxf(mx, __shfl_xor(mx, msk, 32));
        const float mnew = fmaxf(m_r[r], mx);
        alpha[r] = __expf(m_r[r] - mnew);
        m_r[r] = mnew;
        const float p0 = __expf(s0[r] - mnew);
        const float p1 = __expf(s1[r] - mnew);
        s0[r] = p0; s1[r] = p1;
        float rs = p0 + p1;
#pragma unroll
        for (int msk = 1; msk < 16; msk <<= 1) rs += __shfl_xor(rs, msk, 32);
        l_r[r] = l_r[r] * alpha[r] + rs;
    }
#pragma unroll
    for (int r = 0; r < 8; ++r) {
        o0[r] *= alpha[r]; o1[r] *= alpha[r]; o2[r] *= alpha[r]; o3[r] *= alpha[r];
        const int rr = r + 8 * lh;                       // C/D layout row
        myP[rr * 32 + col]      = (_Float16)s0[r];
        myP[rr * 32 + 16 + col] = (_Float16)s1[r];
    }
    __syncthreads();
    {   // P (16x32) as A-fragment from LDS, then O += P V
        const _Float16* pp = myP + (lane & 15) * 32 + lh * 8;
        const v16h pf = combine16(*(const v8h*)pp, *(const v8h*)(pp + 16));
        o0 = wmma16(pf, vf0, o0);
        o1 = wmma16(pf, vf1, o1);
        o2 = wmma16(pf, vf2, o2);
        o3 = wmma16(pf, vf3, o3);
    }
    __syncthreads();
}

// grid = 32 (b,h) * 16 qchunks ; 8 waves/block, each wave owns a 16-row query tile.
__global__ void __launch_bounds__(256) attn_kernel(const _Float16* __restrict__ q,
                                                   const _Float16* __restrict__ k,
                                                   const _Float16* __restrict__ vt,
                                                   _Float16* __restrict__ ao) {
    __shared__ __align__(16) _Float16 sP[8][16 * 32];   // per-wave P staging (8 KB)
    const int lane = threadIdx.x & 31, wave = threadIdx.x >> 5;
    const int col = lane & 15, lh = lane >> 4;
    const int bh = blockIdx.x >> 4;
    const int q0 = (blockIdx.x & 15) * 128 + wave * 16;

    const _Float16* qp  = q  + (size_t)bh * SEQ * HDIM;
    const _Float16* kp  = k  + (size_t)bh * SEQ * HDIM;
    const _Float16* vtp = vt + (size_t)bh * HDIM * SEQ;

    const v16h qf0 = load_a_frag(qp + (size_t)q0 * HDIM, HDIM, 0,  lane);
    const v16h qf1 = load_a_frag(qp + (size_t)q0 * HDIM, HDIM, 32, lane);

    v8f o0 = zero8(), o1 = zero8(), o2 = zero8(), o3 = zero8();
    float m_r[8], l_r[8];
#pragma unroll
    for (int r = 0; r < 8; ++r) { m_r[r] = NEG_INF; l_r[r] = 0.0f; }

    _Float16* myP = sP[wave];

    // prime K fragment set A for m0 = 0
    v16h ka0 = load_b_frag(kp,                     HDIM, 0,  lane);
    v16h ka1 = load_b_frag(kp,                     HDIM, 32, lane);
    v16h ka2 = load_b_frag(kp + (size_t)16 * HDIM, HDIM, 0,  lane);
    v16h ka3 = load_b_frag(kp + (size_t)16 * HDIM, HDIM, 32, lane);
    v16h kb0, kb1, kb2, kb3;

    for (int m0 = 0; m0 < SEQ; m0 += 64) {
        attn_block(kp, vtp, myP, m0,      m0 + 32,              lane, col, lh,
                   qf0, qf1, ka0, ka1, ka2, ka3, kb0, kb1, kb2, kb3,
                   o0, o1, o2, o3, m_r, l_r);
        attn_block(kp, vtp, myP, m0 + 32, (m0 + 64) & (SEQ - 1), lane, col, lh,
                   qf0, qf1, kb0, kb1, kb2, kb3, ka0, ka1, ka2, ka3,
                   o0, o1, o2, o3, m_r, l_r);
    }
    // epilogue: O /= l, write [b, n, h*64 + d] f16
    const int b = bh >> 3, h = bh & 7;
#pragma unroll
    for (int r = 0; r < 8; ++r) {
        const float rl = 1.0f / l_r[r];
        const int rr = r + 8 * lh;
        _Float16* op = ao + ((size_t)b * SEQ + q0 + rr) * C_DIM + h * HDIM;
        op[col]      = (_Float16)(o0[r] * rl);
        op[16 + col] = (_Float16)(o1[r] * rl);
        op[32 + col] = (_Float16)(o2[r] * rl);
        op[48 + col] = (_Float16)(o3[r] * rl);
    }
}

// ------------------------------------------- out = x + attn @ w_proj + b_proj
__global__ void __launch_bounds__(256) proj_gemm(const _Float16* __restrict__ a,
                                                 const _Float16* __restrict__ wT,
                                                 const float* __restrict__ bias,
                                                 const float* __restrict__ x,
                                                 float* __restrict__ y) {
    const int lane = threadIdx.x & 31, wave = threadIdx.x >> 5;
    const int rowBase = blockIdx.y * 256 + wave * 32;
    const int colBase = blockIdx.x * 64;

    v8f acc[2][4];
#pragma unroll
    for (int rb = 0; rb < 2; ++rb)
#pragma unroll
        for (int t = 0; t < 4; ++t) acc[rb][t] = zero8();

    const _Float16* ar0  = a + (size_t)rowBase * C_DIM;
    const _Float16* ar1  = ar0 + (size_t)16 * C_DIM;
    const _Float16* wcol = wT + (size_t)colBase * C_DIM;

    v16h a0A, a1A, bA[4], a0B, a1B, bB[4];
    gemm_load(ar0, ar1, wcol, 0, lane, a0A, a1A, bA);

    for (int k0 = 0; k0 < C_DIM; k0 += 64) {
        gemm_load(ar0, ar1, wcol, k0 + 32, lane, a0B, a1B, bB);
        gemm_mma(acc, a0A, a1A, bA);
        gemm_load(ar0, ar1, wcol, (k0 + 64) & (C_DIM - 1), lane, a0A, a1A, bA);
        gemm_mma(acc, a0B, a1B, bB);
    }

    const int col = lane & 15, lh = lane >> 4;
#pragma unroll
    for (int rb = 0; rb < 2; ++rb) {
#pragma unroll
        for (int t = 0; t < 4; ++t) {
#pragma unroll
            for (int r = 0; r < 8; ++r) {
                const int gm = rowBase + rb * 16 + r + 8 * lh;
                const int gc = colBase + t * 16 + col;
                y[(size_t)gm * C_DIM + gc] = acc[rb][t][r] + bias[gc] + x[(size_t)gm * C_DIM + gc];
            }
        }
    }
}

// ---------------------------------------------------------------- launcher
extern "C" void kernel_launch(void* const* d_in, const int* in_sizes, int n_in,
                              void* d_out, int out_size, void* d_ws, size_t ws_size,
                              hipStream_t stream) {
    const float* x      = (const float*)d_in[0];
    const float* w_qkv  = (const float*)d_in[1];
    const float* w_proj = (const float*)d_in[2];
    const float* b_proj = (const float*)d_in[3];
    const float* gamma  = (const float*)d_in[4];
    const float* beta   = (const float*)d_in[5];
    float* y = (float*)d_out;

    char* ws = (char*)d_ws;
    _Float16* xn     = (_Float16*)(ws);                                  // 8 MB
    _Float16* wqkvT  = (_Float16*)(ws + (size_t)8  * 1024 * 1024);       // 1.5 MB
    _Float16* wprojT = (_Float16*)(ws + (size_t)10 * 1024 * 1024);       // 0.5 MB
    _Float16* qbuf   = (_Float16*)(ws + (size_t)11 * 1024 * 1024);       // 8 MB
    _Float16* kbuf   = (_Float16*)(ws + (size_t)19 * 1024 * 1024);       // 8 MB
    _Float16* vtbuf  = (_Float16*)(ws + (size_t)27 * 1024 * 1024);       // 8 MB
    _Float16* attn   = (_Float16*)(ws + (size_t)35 * 1024 * 1024);       // 8 MB

    ln_kernel<<<ROWS, 256, 0, stream>>>(x, gamma, beta, xn);
    wconv_kernel<<<(C_DIM * N3C + 255) / 256, 256, 0, stream>>>(w_qkv, wqkvT, C_DIM, N3C);
    wconv_kernel<<<(C_DIM * C_DIM + 255) / 256, 256, 0, stream>>>(w_proj, wprojT, C_DIM, C_DIM);
    qkv_gemm<<<dim3(N3C / 64, ROWS / 256), 256, 0, stream>>>(xn, wqkvT, qbuf, kbuf, vtbuf);
    attn_kernel<<<BATCH * HEADS * (SEQ / 128), 256, 0, stream>>>(qbuf, kbuf, vtbuf, attn);
    proj_gemm<<<dim3(C_DIM / 64, ROWS / 256), 256, 0, stream>>>(attn, wprojT, b_proj, x, y);
}
